// LAC_22969485099851
// MI455X (gfx1250) — matrix operand order repaired and checked by reading, not compiled, not run on hardware
//
#include <hip/hip_runtime.h>
#include <hip/hip_bf16.h>
#include <math.h>

// ---------------------------------------------------------------------------
// MI455X (gfx1250) implementation of the LAC block.
// HBM-bound (~0.43 TFLOP vs ~1.1 GB moved @ 23.3 TB/s):
//   * all big GEMMs on v_wmma_f32_16x16x32_bf16 (f32 accumulate)
//   * GEMM-chain intermediates stored directly in bf16 (they are only ever
//     consumed as WMMA A-operands, so this is numerically identical to
//     converting at load time and halves their HBM traffic)
//   * weights pre-packed once into the exact WMMA B-fragment layout
//   * Q/K/V fused into a single N=768 GEMM (x1 read once, not 3x)
//   * epilogues (swish / residual / bias) fused into the GEMM store
//   * global_prefetch_b8 on the A stream; deterministic two-stage reductions
// ---------------------------------------------------------------------------

typedef __bf16 bf16_t;
typedef __attribute__((ext_vector_type(16))) __bf16 bf16x16;
typedef __attribute__((ext_vector_type(8)))  __bf16 bf16x8;
typedef __attribute__((ext_vector_type(8)))  float  f32x8;

__device__ __forceinline__ float sigmoidf_(float x) {
    return 1.0f / (1.0f + __expf(-x));
}

// ---------------------------------------------------------------------------
// Weight pre-pack: fp32 (K x N) -> bf16 fragments in WMMA B-operand layout.
// Fragment (ntile, ktile): lane l (0..31): col = ntile*16 + (l&15),
//   element e (0..15): k = ktile*32 + (l>>4)*8 + (e&7) + (e>>3)*16.
// Source index: W[k*sK + (col/64)*sH + (col%64)]
//   normal matrix : sK = N,  sH = 64     (=> W[k*N + col])
//   head-stacked  : sK = 64, sH = K*64   (=> W[n*K*64 + k*64 + d])
// ---------------------------------------------------------------------------
__global__ __launch_bounds__(256)
void pack_b_kernel(const float* __restrict__ Wsrc, bf16_t* __restrict__ dst,
                   int K, int N, int sK, int sH)
{
    int tid = blockIdx.x * 256 + threadIdx.x;
    int kTiles = K >> 5;
    int nTiles = N >> 4;
    int total = nTiles * kTiles * 32;
    if (tid >= total) return;
    int lane = tid & 31;
    int kt = (tid >> 5) % kTiles;
    int nt = (tid >> 5) / kTiles;
    int half = lane >> 4, l15 = lane & 15;
    int col = nt * 16 + l15;
    int k0 = kt * 32 + half * 8;
    int chh = col >> 6, cl = col & 63;
    bf16_t* d = dst + (size_t)tid * 16;
#pragma unroll
    for (int e = 0; e < 16; ++e)
        d[e] = (bf16_t)Wsrc[(size_t)(k0 + (e & 7) + ((e >> 3) << 4)) * sK + (size_t)chh * sH + cl];
}

// concat 3 x 256-float bias vectors into one 768-float vector
__global__ __launch_bounds__(256)
void concat3_kernel(const float* __restrict__ a, const float* __restrict__ b,
                    const float* __restrict__ c, float* __restrict__ o)
{
    int t = blockIdx.x * 256 + threadIdx.x;   // grid = 3 blocks
    const float* s = (t < 256) ? a : (t < 512) ? b : c;
    o[t] = s[t & 255];
}

// ---------------------------------------------------------------------------
// WMMA GEMM: C(MxN) = epilogue( A(MxK) * Bpacked(KxN,bf16) )
// Block: 256 threads = 8 waves as 2(M) x 4(N); wave = 16 rows x 64 cols
//        (4 x v_wmma_f32_16x16x32_bf16 per k-step, shared A fragment).
// ABF: A operand stored as bf16 (two 16B loads/frag) vs fp32 (load+cvt).
// OBF: output stored as bf16 (next GEMM's A operand) vs fp32.
// EPI: 0 = store, 1 = swish, 2 = res + alpha*v, 3 = v + colbias[c],
//      4 = res + v + colbias[c]
// ---------------------------------------------------------------------------
template<int EPI, int ABF, int OBF>
__global__ __launch_bounds__(256)
void wmma_gemm(const void* __restrict__ Av, const bf16_t* __restrict__ Bp,
               void* __restrict__ Cv, const float* __restrict__ res,
               const float* __restrict__ colbias,
               int M, int N, int K, float alpha)
{
    const int lane = threadIdx.x & 31;
    const int wave = threadIdx.x >> 5;
    const int half = lane >> 4;
    const int l15  = lane & 15;

    const int mWave = blockIdx.y * 32 + (wave >> 2) * 16;
    const int nWave = blockIdx.x * 256 + (wave & 3) * 64;

    const int kTiles = K >> 5;
    const float*  __restrict__ arowF =
        (const float*)Av + (ABF ? 0 : (size_t)(mWave + l15) * K);
    const bf16_t* __restrict__ arowB =
        (const bf16_t*)Av + (ABF ? (size_t)(mWave + l15) * K : 0);
    const bf16_t* __restrict__ bcol =
        Bp + ((size_t)(nWave >> 4) * kTiles) * 512 + lane * 16;
    const size_t nStride = (size_t)kTiles * 512;   // bf16 elems per n-tile

    f32x8 acc0 = {0.f,0.f,0.f,0.f,0.f,0.f,0.f,0.f};
    f32x8 acc1 = acc0, acc2 = acc0, acc3 = acc0;

    for (int kt = 0; kt < kTiles; ++kt) {
        bf16x16 a;
        if (ABF) {
            // bf16 A: ISA 16-bit A 16x32 layout = two contiguous 8-elem runs
            const bf16_t* p0 = arowB + (kt << 5) + half * 8;
            __builtin_prefetch(p0 + 128, 0, 0);          // 4 k-tiles ahead
            bf16x8 lo = *(const bf16x8*)(p0);
            bf16x8 hi = *(const bf16x8*)(p0 + 16);
            a = __builtin_shufflevector(lo, hi, 0, 1, 2, 3, 4, 5, 6, 7,
                                        8, 9, 10, 11, 12, 13, 14, 15);
        } else {
            const float* p0 = arowF + (kt << 5) + half * 8;
            __builtin_prefetch(p0 + 128, 0, 0);
#pragma unroll
            for (int e = 0; e < 8; ++e) {
                a[e]     = (bf16_t)p0[e];
                a[e + 8] = (bf16_t)p0[e + 16];
            }
        }
        const bf16_t* bb = bcol + (size_t)kt * 512;
        bf16x16 b0 = *(const bf16x16*)(bb);
        bf16x16 b1 = *(const bf16x16*)(bb + nStride);
        bf16x16 b2 = *(const bf16x16*)(bb + 2 * nStride);
        bf16x16 b3 = *(const bf16x16*)(bb + 3 * nStride);
        acc0 = __builtin_amdgcn_wmma_f32_16x16x32_bf16(false, a, false, b0, (short)0, acc0, false, false);
        acc1 = __builtin_amdgcn_wmma_f32_16x16x32_bf16(false, a, false, b1, (short)0, acc1, false, false);
        acc2 = __builtin_amdgcn_wmma_f32_16x16x32_bf16(false, a, false, b2, (short)0, acc2, false, false);
        acc3 = __builtin_amdgcn_wmma_f32_16x16x32_bf16(false, a, false, b3, (short)0, acc3, false, false);
    }

    float*  Cf = (float*)Cv;
    bf16_t* Cb = (bf16_t*)Cv;
    f32x8 accs[4] = {acc0, acc1, acc2, acc3};
#pragma unroll
    for (int t = 0; t < 4; ++t) {
        const int col = nWave + t * 16 + l15;
#pragma unroll
        for (int r = 0; r < 8; ++r) {
            const int orow = mWave + half * 8 + r;       // C/D VGPR layout
            const size_t idx = (size_t)orow * N + col;
            float v = accs[t][r];
            if (EPI == 1)      v = v * sigmoidf_(v);
            else if (EPI == 2) v = res[idx] + alpha * v;
            else if (EPI == 3) v = v + colbias[col];
            else if (EPI == 4) v = res[idx] + v + colbias[col];
            if (OBF) Cb[idx] = (bf16_t)v;
            else     Cf[idx] = v;
        }
    }
}

// ---------------------------------------------------------------------------
// Column softmax over H=1024. Input stride xs (768 for fused QKV buffer),
// output stride 256. grid = B*256 (one block per (batch, column)).
// ---------------------------------------------------------------------------
__global__ __launch_bounds__(256)
void softmax_col(const float* __restrict__ X, int xs,
                 float* __restrict__ Y, float scale)
{
    int col = blockIdx.x & 255;
    int b   = blockIdx.x >> 8;
    const float* x = X + (size_t)b * 1024 * xs + col;
    float*       y = Y + (size_t)b * 1024 * 256 + col;
    int t = threadIdx.x;
    float v[4];
    float mx = -3.4e38f;
#pragma unroll
    for (int i = 0; i < 4; ++i) {
        v[i] = x[(size_t)(t + i * 256) * xs] * scale;
        mx = fmaxf(mx, v[i]);
    }
    __shared__ float red[256];
    red[t] = mx; __syncthreads();
    for (int s = 128; s > 0; s >>= 1) {
        if (t < s) red[t] = fmaxf(red[t], red[t + s]);
        __syncthreads();
    }
    mx = red[0]; __syncthreads();
    float sm = 0.f;
#pragma unroll
    for (int i = 0; i < 4; ++i) { v[i] = __expf(v[i] - mx); sm += v[i]; }
    red[t] = sm; __syncthreads();
    for (int s = 128; s > 0; s >>= 1) {
        if (t < s) red[t] += red[t + s];
        __syncthreads();
    }
    float invs = 1.f / red[0];
#pragma unroll
    for (int i = 0; i < 4; ++i)
        y[(size_t)(t + i * 256) * 256] = v[i] * invs;
}

// ---------------------------------------------------------------------------
// Bm[b,n] = Ksm[b,:,n]^T @ V[b,:,n]  : (64x64) with K=1024. grid = B*4.
// Ksm stride 256 (compact), V stride vs (768 inside fused QKV buffer).
// ---------------------------------------------------------------------------
__global__ __launch_bounds__(256)
void bm_kernel(const float* __restrict__ Ksm, const float* __restrict__ V,
               int vs, float* __restrict__ Bm)
{
    int n = blockIdx.x & 3, b = blockIdx.x >> 2;
    __shared__ float Ks[64][65];
    __shared__ float Vs[64][65];
    int t = threadIdx.x;
    int d  = t >> 2;            // 0..63
    int e0 = (t & 3) * 16;      // 16-wide e strip
    float acc[16];
#pragma unroll
    for (int i = 0; i < 16; ++i) acc[i] = 0.f;
    const size_t kbase = (size_t)b * 1024 * 256 + n * 64;
    const size_t vbase = (size_t)b * 1024 * vs + n * 64;
    for (int h0 = 0; h0 < 1024; h0 += 64) {
        __syncthreads();
        for (int i = t; i < 64 * 64; i += 256) {
            int hh = i >> 6, dd = i & 63;
            Ks[hh][dd] = Ksm[kbase + (size_t)(h0 + hh) * 256 + dd];
            Vs[hh][dd] = V[vbase + (size_t)(h0 + hh) * vs + dd];
        }
        __syncthreads();
        for (int h = 0; h < 64; ++h) {
            float kd = Ks[h][d];
#pragma unroll
            for (int j = 0; j < 16; ++j) acc[j] += kd * Vs[h][e0 + j];
        }
    }
    float* outp = Bm + (size_t)(b * 4 + n) * 4096 + d * 64 + e0;
#pragma unroll
    for (int j = 0; j < 16; ++j) outp[j] = acc[j];
}

// ---------------------------------------------------------------------------
// Z[b,:,n] = Asm[b,:,n] @ Bm[b,n] : (1024x64)@(64x64). grid = B*4*64.
// Z written in bf16 (it is only consumed as the wo-GEMM A operand).
// ---------------------------------------------------------------------------
__global__ __launch_bounds__(256)
void z_kernel(const float* __restrict__ Asm, const float* __restrict__ Bm,
              bf16_t* __restrict__ Z)
{
    int hb = blockIdx.x & 63;
    int n  = (blockIdx.x >> 6) & 3;
    int b  = blockIdx.x >> 8;
    __shared__ float Bs[64][64];
    __shared__ float As[16][64];
    int t = threadIdx.x;
    size_t bmbase = (size_t)(b * 4 + n) * 4096;
    for (int i = t; i < 4096; i += 256) ((float*)Bs)[i] = Bm[bmbase + i];
    size_t abase = ((size_t)b * 1024 + hb * 16) * 256 + n * 64;
    for (int i = t; i < 16 * 64; i += 256) {
        int r = i >> 6, c = i & 63;
        As[r][c] = Asm[abase + (size_t)r * 256 + c];
    }
    __syncthreads();
    int e = t & 63, i0 = t >> 6;   // 4 rows per thread
    float acc[4] = {0.f, 0.f, 0.f, 0.f};
    for (int d = 0; d < 64; ++d) {
        float bv = Bs[d][e];
#pragma unroll
        for (int r = 0; r < 4; ++r) acc[r] += As[i0 + r * 4][d] * bv;
    }
#pragma unroll
    for (int r = 0; r < 4; ++r)
        Z[abase + (size_t)(i0 + r * 4) * 256 + e] = (bf16_t)acc[r];
}

// ---------------------------------------------------------------------------
// Deterministic two-stage per-batch mean/var (no atomics).
// ---------------------------------------------------------------------------
__global__ __launch_bounds__(256)
void reduce_partial(const float* __restrict__ X, float* __restrict__ part,
                    int chunkElems, int chunks)
{
    const int b = blockIdx.y, c = blockIdx.x;
    const float* x = X + (size_t)b * chunks * chunkElems + (size_t)c * chunkElems;
    float s = 0.f, q = 0.f;
    for (int i = threadIdx.x; i < chunkElems; i += 256) {
        float v = x[i]; s += v; q += v * v;
    }
    __shared__ float rs[256], rq[256];
    rs[threadIdx.x] = s; rq[threadIdx.x] = q;
    __syncthreads();
    for (int st = 128; st > 0; st >>= 1) {
        if (threadIdx.x < st) {
            rs[threadIdx.x] += rs[threadIdx.x + st];
            rq[threadIdx.x] += rq[threadIdx.x + st];
        }
        __syncthreads();
    }
    if (threadIdx.x == 0) {
        part[(b * chunks + c) * 2]     = rs[0];
        part[(b * chunks + c) * 2 + 1] = rq[0];
    }
}

__global__ __launch_bounds__(64)
void reduce_final(const float* __restrict__ part, float* __restrict__ mu,
                  float* __restrict__ inv, int chunks, float invN)
{
    const int b = blockIdx.x;
    __shared__ float rs[64], rq[64];
    float s = 0.f, q = 0.f;
    for (int i = threadIdx.x; i < chunks; i += 64) {
        s += part[(b * chunks + i) * 2];
        q += part[(b * chunks + i) * 2 + 1];
    }
    rs[threadIdx.x] = s; rq[threadIdx.x] = q;
    __syncthreads();
    for (int st = 32; st > 0; st >>= 1) {
        if (threadIdx.x < st) {
            rs[threadIdx.x] += rs[threadIdx.x + st];
            rq[threadIdx.x] += rq[threadIdx.x + st];
        }
        __syncthreads();
    }
    if (threadIdx.x == 0) {
        float m = rs[0] * invN;
        float var = rq[0] * invN - m * m;
        mu[b]  = m;
        inv[b] = rsqrtf(var + 1e-5f);
    }
}

// ---------------------------------------------------------------------------
// Fused conv module (C=1): x3 = x2 + pw2(swish(bn(dw2(conv3(glu(pw1(ln(x2))))))))
// ---------------------------------------------------------------------------
__global__ __launch_bounds__(256)
void conv_module_k(const float* __restrict__ x2, const float* __restrict__ mu,
                   const float* __restrict__ inv,
                   const float* __restrict__ g, const float* __restrict__ bet,
                   const float* __restrict__ pw1w, const float* __restrict__ pw1b,
                   const float* __restrict__ dw1w, const float* __restrict__ dw1b,
                   const float* __restrict__ dw2w, const float* __restrict__ dw2b,
                   const float* __restrict__ bng, const float* __restrict__ bnb,
                   const float* __restrict__ bnm, const float* __restrict__ bnv,
                   const float* __restrict__ pw2w, const float* __restrict__ pw2b,
                   float* __restrict__ x3)
{
    const int Wd = 512;
    size_t idx = (size_t)blockIdx.x * 256 + threadIdx.x;   // over B*H*W
    int w = (int)(idx % Wd);
    size_t bh = idx / Wd;
    int h = (int)(bh & 1023);
    int b = (int)(bh >> 10);
    int h2 = h & 511;          // GLU folds H=1024 -> 512, dw2 unfolds
    int ch = h >> 9;           // dw2 channel this output row came from
    float m = mu[b], iv = inv[b];
    float p1w = pw1w[0], p1b = pw1b[0];
    const float* xb = x2 + (size_t)b * (1024 * 512);
    float glu[3];
#pragma unroll
    for (int j = 0; j < 3; ++j) {
        int wj = w + j - 1;                 // zero-padded width-3 stencil
        float gv = 0.f;
        if (wj >= 0 && wj < Wd) {
            int iu = h2 * Wd + wj;
            int ig = (h2 + 512) * Wd + wj;
            float u  = ((xb[iu] - m) * iv * g[iu] + bet[iu]) * p1w + p1b;
            float gt = ((xb[ig] - m) * iv * g[ig] + bet[ig]) * p1w + p1b;
            gv = u * sigmoidf_(gt);
        }
        glu[j] = gv;
    }
    float v = dw1w[0] * glu[0] + dw1w[1] * glu[1] + dw1w[2] * glu[2] + dw1b[0];
    v = v * dw2w[ch] + dw2b[ch];
    v = (v - bnm[0]) * rsqrtf(bnv[0] + 1e-5f) * bng[0] + bnb[0];
    v = v * sigmoidf_(v);
    v = v * pw2w[0] + pw2b[0];
    x3[idx] = x2[idx] + v;
}

// ---------------------------------------------------------------------------
// out = inputs + LN_{C,H,W}(x4)
// ---------------------------------------------------------------------------
__global__ __launch_bounds__(256)
void final_ln(const float* __restrict__ x0, const float* __restrict__ x4,
              const float* __restrict__ mu, const float* __restrict__ inv,
              const float* __restrict__ g, const float* __restrict__ bet,
              float* __restrict__ out)
{
    size_t idx = (size_t)blockIdx.x * 256 + threadIdx.x;
    const size_t per = (size_t)1024 * 512;
    int b = (int)(idx / per);
    size_t r = idx % per;
    float v = (x4[idx] - mu[b]) * inv[b] * g[r] + bet[r];
    out[idx] = x0[idx] + v;
}

// ---------------------------------------------------------------------------
// Host driver
// ---------------------------------------------------------------------------
extern "C" void kernel_launch(void* const* d_in, const int* in_sizes, int n_in,
                              void* d_out, int out_size, void* d_ws, size_t ws_size,
                              hipStream_t stream)
{
    (void)in_sizes; (void)n_in; (void)out_size; (void)ws_size;

    const int B = 32, H = 1024, Wd = 512;
    const int M = B * H;                       // 32768
    const float invdq = 0.35355339059327373f;  // 1 / 64^0.25

    const float* x0   = (const float*)d_in[0];
    const float* l1e1 = (const float*)d_in[1];
    const float* l1d1 = (const float*)d_in[2];
    const float* l1e2 = (const float*)d_in[3];
    const float* l1d2 = (const float*)d_in[4];
    const float* wq   = (const float*)d_in[5];
    const float* bq   = (const float*)d_in[6];
    const float* wk   = (const float*)d_in[7];
    const float* bk   = (const float*)d_in[8];
    const float* wv   = (const float*)d_in[9];
    const float* bv   = (const float*)d_in[10];
    const float* wo   = (const float*)d_in[11];
    const float* bo   = (const float*)d_in[12];
    const float* ln1g = (const float*)d_in[13];
    const float* ln1b = (const float*)d_in[14];
    const float* pw1w = (const float*)d_in[15];
    const float* pw1b = (const float*)d_in[16];
    const float* dw1w = (const float*)d_in[17];
    const float* dw1b = (const float*)d_in[18];
    const float* dw2w = (const float*)d_in[19];
    const float* dw2b = (const float*)d_in[20];
    const float* bng  = (const float*)d_in[21];
    const float* bnb  = (const float*)d_in[22];
    const float* bnm  = (const float*)d_in[23];
    const float* bnv  = (const float*)d_in[24];
    const float* pw2w = (const float*)d_in[25];
    const float* pw2b = (const float*)d_in[26];
    const float* l2e1 = (const float*)d_in[27];
    const float* l2d1 = (const float*)d_in[28];
    const float* l2e2 = (const float*)d_in[29];
    const float* l2d2 = (const float*)d_in[30];
    const float* lnfg = (const float*)d_in[31];
    const float* lnfb = (const float*)d_in[32];
    float* out = (float*)d_out;

    // ---- workspace carving (all offsets 256B aligned) ----
    char* ws = (char*)d_ws;
    size_t off = 0;
    auto alloc = [&](size_t bytes) -> void* {
        void* p = ws + off;
        off += (bytes + 255) & ~(size_t)255;
        return p;
    };
    float* bufA = (float*)alloc((size_t)M * 1024 * 4);   // 128 MiB ping
    float* bufB = (float*)alloc((size_t)M * 1024 * 4);   // 128 MiB pong
    float* S1   = (float*)alloc((size_t)M * 512 * 4);    // x1 -> x3 (fp32)
    float* S2   = (float*)alloc((size_t)M * 512 * 4);    // x2 -> x4 (fp32)
    bf16_t* e1p  = (bf16_t*)alloc((size_t)512 * 1024 * 2);
    bf16_t* d1p  = (bf16_t*)alloc((size_t)1024 * 1024 * 2);
    bf16_t* e2p  = (bf16_t*)alloc((size_t)1024 * 1024 * 2);
    bf16_t* d2p  = (bf16_t*)alloc((size_t)1024 * 512 * 2);
    bf16_t* e1p2 = (bf16_t*)alloc((size_t)512 * 1024 * 2);
    bf16_t* d1p2 = (bf16_t*)alloc((size_t)1024 * 1024 * 2);
    bf16_t* e2p2 = (bf16_t*)alloc((size_t)1024 * 1024 * 2);
    bf16_t* d2p2 = (bf16_t*)alloc((size_t)1024 * 512 * 2);
    bf16_t* qkvp = (bf16_t*)alloc((size_t)512 * 768 * 2);  // wq|wk|wv packed
    bf16_t* wop  = (bf16_t*)alloc((size_t)256 * 512 * 2);
    float* qkvb = (float*)alloc(768 * 4);                  // bq|bk|bv
    float* part = (float*)alloc((size_t)B * 64 * 2 * 4);
    float* mu1  = (float*)alloc(B * 4);
    float* inv1 = (float*)alloc(B * 4);
    float* mu2  = (float*)alloc(B * 4);
    float* inv2 = (float*)alloc(B * 4);

    // GEMM-chain intermediates live in bufA/bufB reinterpreted as bf16.
    bf16_t* t1 = (bf16_t*)bufA;                 // M x 1024 bf16
    bf16_t* t2 = (bf16_t*)bufB;                 // M x 1024 bf16
    bf16_t* t3 = (bf16_t*)bufA;                 // overwrites t1

    // MHLA scratch carved from the (then free) big buffers:
    float*  QKV = bufA;                               // M x 768 fp32 (96 MiB)
    bf16_t* Zb  = (bf16_t*)(bufA + (size_t)M * 768);  // M x 256 bf16 (16 MiB)
    float*  Asm = bufB;                               // M x 256 fp32
    float*  Ksm = bufB + (size_t)M * 256;             // M x 256 fp32
    float*  Bmb = bufB + (size_t)M * 512;             // 32*4*64*64 fp32

    dim3 blk(256);
    auto packLaunch = [&](const float* src, bf16_t* dst, int K, int N, int sK, int sH) {
        int total = (N >> 4) * (K >> 5) * 32;
        pack_b_kernel<<<(total + 255) / 256, blk, 0, stream>>>(src, dst, K, N, sK, sH);
    };
    auto gemmGrid = [&](int N) { return dim3(N / 256, M / 32); };

    // ---- pack all weights to bf16 WMMA fragments ----
    packLaunch(l1e1, e1p, 512, 1024, 1024, 64);
    packLaunch(l1d1, d1p, 1024, 1024, 1024, 64);
    packLaunch(l1e2, e2p, 1024, 1024, 1024, 64);
    packLaunch(l1d2, d2p, 1024, 512, 512, 64);
    packLaunch(l2e1, e1p2, 512, 1024, 1024, 64);
    packLaunch(l2d1, d1p2, 1024, 1024, 1024, 64);
    packLaunch(l2e2, e2p2, 1024, 1024, 1024, 64);
    packLaunch(l2d2, d2p2, 1024, 512, 512, 64);
    // fused QKV weight: packed layout is n-tile-major, so the three head-
    // stacked weights concatenate by writing adjacent n-tile ranges.
    packLaunch(wq, qkvp,                        512, 256, 64, 512 * 64);
    packLaunch(wk, qkvp + (size_t)512 * 256,    512, 256, 64, 512 * 64);
    packLaunch(wv, qkvp + (size_t)512 * 512,    512, 256, 64, 512 * 64);
    packLaunch(wo, wop, 256, 512, 512, 64);
    concat3_kernel<<<3, blk, 0, stream>>>(bq, bk, bv, qkvb);

    // ---- x1 = x0 + 0.5 * lffn1(x0) ----
    wmma_gemm<0,0,1><<<gemmGrid(1024), blk, 0, stream>>>(x0, e1p, t1, nullptr, nullptr, M, 1024, 512, 0.f);
    wmma_gemm<1,1,1><<<gemmGrid(1024), blk, 0, stream>>>(t1, d1p, t2, nullptr, nullptr, M, 1024, 1024, 0.f); // swish
    wmma_gemm<0,1,1><<<gemmGrid(1024), blk, 0, stream>>>(t2, e2p, t3, nullptr, nullptr, M, 1024, 1024, 0.f);
    wmma_gemm<2,1,0><<<gemmGrid(512),  blk, 0, stream>>>(t3, d2p, S1, x0,      nullptr, M, 512, 1024, 0.5f);

    // ---- x2 = x1 + MHLA(x1) ----
    wmma_gemm<3,0,0><<<gemmGrid(768), blk, 0, stream>>>(S1, qkvp, QKV, nullptr, qkvb, M, 768, 512, 0.f);
    softmax_col<<<B * 256, blk, 0, stream>>>(QKV + 0,   768, Asm, invdq);   // Q
    softmax_col<<<B * 256, blk, 0, stream>>>(QKV + 256, 768, Ksm, invdq);   // K
    bm_kernel<<<B * 4, blk, 0, stream>>>(Ksm, QKV + 512, 768, Bmb);         // V
    z_kernel<<<B * 4 * 64, blk, 0, stream>>>(Asm, Bmb, Zb);
    wmma_gemm<4,1,0><<<gemmGrid(512), blk, 0, stream>>>(Zb, wop, S2, S1, bo, M, 512, 256, 0.f);

    // ---- x3 = x2 + conv_module(x2) ----
    reduce_partial<<<dim3(64, B), blk, 0, stream>>>(S2, part, (H * Wd) / 64, 64);
    reduce_final<<<B, 64, 0, stream>>>(part, mu1, inv1, 64, 1.f / (float)(H * Wd));
    conv_module_k<<<(M * Wd) / 256, blk, 0, stream>>>(
        S2, mu1, inv1, ln1g, ln1b, pw1w, pw1b, dw1w, dw1b, dw2w, dw2b,
        bng, bnb, bnm, bnv, pw2w, pw2b, S1);

    // ---- x4 = x3 + 0.5 * lffn2(x3) ----
    wmma_gemm<0,0,1><<<gemmGrid(1024), blk, 0, stream>>>(S1, e1p2, t1, nullptr, nullptr, M, 1024, 512, 0.f);
    wmma_gemm<1,1,1><<<gemmGrid(1024), blk, 0, stream>>>(t1, d1p2, t2, nullptr, nullptr, M, 1024, 1024, 0.f);
    wmma_gemm<0,1,1><<<gemmGrid(1024), blk, 0, stream>>>(t2, e2p2, t3, nullptr, nullptr, M, 1024, 1024, 0.f);
    wmma_gemm<2,1,0><<<gemmGrid(512),  blk, 0, stream>>>(t3, d2p2, S2, S1,      nullptr, M, 512, 1024, 0.5f);

    // ---- out = x0 + LN(x4) ----
    reduce_partial<<<dim3(64, B), blk, 0, stream>>>(S2, part, (H * Wd) / 64, 64);
    reduce_final<<<B, 64, 0, stream>>>(part, mu2, inv2, 64, 1.f / (float)(H * Wd));
    final_ln<<<(M * Wd) / 256, blk, 0, stream>>>(x0, S2, mu2, inv2, lnfg, lnfb, out);
}
